// Attention_83056077570791
// MI455X (gfx1250) — matrix operand order repaired and checked by reading, not compiled, not run on hardware
//
#include <hip/hip_runtime.h>

// MI455X (gfx1250) wave32 WMMA implementation.
// Workspace layout (needs ~25 MB):
//   xnT  [4][4096][256] f16   8388608 B
//   wq   [384][256]    f16     196608 B   (Q rows pre-scaled by log2e/sqrt(32))
//   wo   [256][128]    f16      65536 B
//   Q    [4][4][4096][32] f16 4194304 B
//   K    [4][4][4096][32] f16 4194304 B
//   VT   [4][4][32][4096] f16 4194304 B
//   O    [4][4096][128]   f16 4194304 B

typedef __attribute__((ext_vector_type(16))) _Float16 v16h;
typedef __attribute__((ext_vector_type(8)))  float    v8f;

#define BATCH 4
#define C_IN 256
#define HW   4096
#define HID  128
#define NH   4
#define HD   32
#define TT   HW

union Frag {
    v16h v;
    uint4 q[2];
    unsigned w[8];
    _Float16 h[16];
};

// A-fragment (16x32 f16, MxK) from row-major src with leading dim ld.
// Lane m=l&15 holds K=16hs..16hs+15 contiguously (VGPR0-3: 16hs+0..7,
// VGPR4-7: 16hs+8..15) with hs=l>>4.
__device__ __forceinline__ Frag load_a(const _Float16* src, int ld) {
    int lane = threadIdx.x & 31;
    int m = lane & 15, hs = lane >> 4;
    const _Float16* row = src + (size_t)m * ld;
    Frag f;
    f.q[0] = *(const uint4*)(row + 16 * hs);
    f.q[1] = *(const uint4*)(row + 16 * hs + 8);
    return f;
}

// B-fragment (32x16 f16, KxN): b[i] = B[16*hs + i][n].  Here B[k][n] is read
// from the *transposed* storage srcT (row n, col k): 16 contiguous halves.
__device__ __forceinline__ Frag load_b(const _Float16* srcT, int ld) {
    int lane = threadIdx.x & 31;
    int n = lane & 15, hs = lane >> 4;
    const _Float16* p = srcT + (size_t)n * ld + 16 * hs;
    Frag f;
    f.q[0] = *(const uint4*)(p);
    f.q[1] = *(const uint4*)(p + 8);
    return f;
}

__device__ __forceinline__ v8f wmma16(const Frag& a, const Frag& b, v8f c) {
    return __builtin_amdgcn_wmma_f32_16x16x32_f16(false, a.v, false, b.v,
                                                  (short)0, c, false, false);
}

__device__ __forceinline__ unsigned pack2(float a, float b) {
    union { _Float16 h[2]; unsigned u; } p;
    p.h[0] = (_Float16)a; p.h[1] = (_Float16)b;
    return p.u;
}

// ---------------- Kernel 1: GroupNorm -> xnT f16 [b][t][c] -----------------
__global__ void k_groupnorm(const float* __restrict__ x,
                            const float* __restrict__ nw,
                            const float* __restrict__ nb,
                            _Float16* __restrict__ xnT) {
    int b = blockIdx.x >> 3, g = blockIdx.x & 7;
    const float* base = x + ((size_t)(b * C_IN + g * 32)) * HW;
    int tid = threadIdx.x;
    float s = 0.f, ss = 0.f;
    for (int i = tid; i < 32 * HW; i += 256) {
        float v = base[i]; s += v; ss += v * v;
    }
    __shared__ float rs[256], rss[256];
    __shared__ float smean, srstd;
    rs[tid] = s; rss[tid] = ss;
    __syncthreads();
    for (int off = 128; off > 0; off >>= 1) {
        if (tid < off) { rs[tid] += rs[tid + off]; rss[tid] += rss[tid + off]; }
        __syncthreads();
    }
    if (tid == 0) {
        float mean = rs[0] * (1.f / (32.f * HW));
        float var  = rss[0] * (1.f / (32.f * HW)) - mean * mean;
        smean = mean; srstd = rsqrtf(var + 1e-5f);
    }
    __syncthreads();
    float mean = smean, rstd = srstd;
    float wv[32], bv[32];
#pragma unroll
    for (int c = 0; c < 32; c++) { wv[c] = nw[g * 32 + c]; bv[c] = nb[g * 32 + c]; }
    for (int t = tid; t < HW; t += 256) {
        union { _Float16 h[32]; uint4 q[4]; } out;
#pragma unroll
        for (int c = 0; c < 32; c++) {
            float v = base[(size_t)c * HW + t];
            out.h[c] = (_Float16)((v - mean) * rstd * wv[c] + bv[c]);
        }
        uint4* dst = (uint4*)(xnT + ((size_t)(b * HW + t)) * C_IN + g * 32);
        dst[0] = out.q[0]; dst[1] = out.q[1]; dst[2] = out.q[2]; dst[3] = out.q[3];
    }
}

// ---------------- Kernel 2: weights -> f16 (fold log2e*scale into Q rows) --
__global__ void k_cvt_w(const float* __restrict__ qkv_w,
                        const float* __restrict__ out_w,
                        _Float16* __restrict__ wq, _Float16* __restrict__ wo) {
    int i = blockIdx.x * 256 + threadIdx.x;   // grid covers exactly 131072
    const float SC = 0.25503485f;             // log2(e) / sqrt(32)
    if (i < 384 * C_IN) {
        float v = qkv_w[i];
        if (i < 128 * C_IN) v *= SC;
        wq[i] = (_Float16)v;
    } else {
        int j = i - 384 * C_IN;
        wo[j] = (_Float16)out_w[j];
    }
}

// ---------------- Kernel 3: QKV GEMM (384x4096 per batch, K=256) -----------
__global__ void k_qkv(const _Float16* __restrict__ wq,
                      const _Float16* __restrict__ xnT,
                      _Float16* __restrict__ Q, _Float16* __restrict__ K,
                      _Float16* __restrict__ VT) {
    int task = blockIdx.x * 8 + (threadIdx.x >> 5);   // 24576 wave-tasks
    int b   = task / (256 * 24);
    int rem = task % (256 * 24);
    int ot  = rem >> 8;
    int tt  = rem & 255;
    int obase = ot * 16, tbase = tt * 16;
    const _Float16* arow = wq + (size_t)obase * C_IN;
    const _Float16* brow = xnT + ((size_t)(b * HW + tbase)) * C_IN;
    v8f acc = {};
#pragma unroll
    for (int kc = 0; kc < C_IN; kc += 32) {
        Frag a  = load_a(arow + kc, C_IN);
        Frag bb = load_b(brow + kc, C_IN);
        acc = wmma16(a, bb, acc);
    }
    int lane = threadIdx.x & 31;
    int n = lane & 15, hs = lane >> 4;
    int t = tbase + n;
    int region = obase >> 7;          // 0=Q, 1=K, 2=V (tile never straddles)
    int olocal = obase & 127;
    int h = olocal >> 5;
    int dlo = (olocal & 31) + 8 * hs; // first of this lane's 8 d values
    if (region < 2) {
        union { unsigned w[4]; uint4 q; } pk;
#pragma unroll
        for (int j = 0; j < 4; j++) pk.w[j] = pack2(acc[2 * j], acc[2 * j + 1]);
        _Float16* dst = (region == 0 ? Q : K);
        *(uint4*)(dst + ((size_t)(b * NH + h) * TT + t) * HD + dlo) = pk.q;
    } else {
#pragma unroll
        for (int r = 0; r < 8; r++)
            VT[((size_t)(b * NH + h) * HD + dlo + r) * TT + t] = (_Float16)acc[r];
    }
}

// ---------------- Kernel 4: flash attention (transposed), one q-tile/wave --
__global__ void k_attn(const _Float16* __restrict__ Q,
                       const _Float16* __restrict__ K,
                       const _Float16* __restrict__ VT,
                       _Float16* __restrict__ O) {
    int task = blockIdx.x * 8 + (threadIdx.x >> 5);   // 4096 wave-tasks
    int qt = task & 255;
    int h  = (task >> 8) & 3;
    int b  = task >> 10;
    int lane  = threadIdx.x & 31;
    int qlane = lane & 15, hs = lane >> 4;
    size_t bh = (size_t)(b * NH + h);
    const _Float16* Qp = Q  + (bh * TT + qt * 16) * HD;
    const _Float16* Kp = K  + bh * TT * HD;
    const _Float16* Vp = VT + bh * HD * TT;

    Frag qf = load_b(Qp, HD);          // B = Q^T (d x 16q), contiguous rows
    v8f olo = {}, ohi = {};
    float mrun = -__builtin_inff(), lrun = 0.f;

    for (int kj = 0; kj < TT; kj += 32) {
        Frag ka0 = load_a(Kp + (size_t)kj * HD, HD);
        Frag ka1 = load_a(Kp + (size_t)(kj + 16) * HD, HD);
        v8f z = {};
        v8f s0 = wmma16(ka0, qf, z);   // S^T rows = keys, cols = queries
        v8f s1 = wmma16(ka1, qf, z);   // (log2-domain: scale folded into wq)

        float cm = s0[0];
#pragma unroll
        for (int r = 1; r < 8; r++) cm = fmaxf(cm, s0[r]);
#pragma unroll
        for (int r = 0; r < 8; r++) cm = fmaxf(cm, s1[r]);
        cm = fmaxf(cm, __shfl_xor(cm, 16, 32));
        float mnew  = fmaxf(mrun, cm);
        float alpha = exp2f(mrun - mnew);
        mrun = mnew;

        float p0[8], p1[8], rsum = 0.f;
#pragma unroll
        for (int r = 0; r < 8; r++) { p0[r] = exp2f(s0[r] - mnew); rsum += p0[r]; }
#pragma unroll
        for (int r = 0; r < 8; r++) { p1[r] = exp2f(s1[r] - mnew); rsum += p1[r]; }
        rsum += __shfl_xor(rsum, 16, 32);
        lrun = lrun * alpha + rsum;
#pragma unroll
        for (int r = 0; r < 8; r++) { olo[r] *= alpha; ohi[r] *= alpha; }

        // S^T D-layout -> P B-fragment (b[i] = P[16hs+i][q]) via bpermute
        unsigned d0[4], d1[4];
#pragma unroll
        for (int j = 0; j < 4; j++) {
            d0[j] = pack2(p0[2 * j], p0[2 * j + 1]);
            d1[j] = pack2(p1[2 * j], p1[2 * j + 1]);
        }
        Frag pf;
#pragma unroll
        for (int j = 0; j < 4; j++) {
            unsigned a0 = __shfl(d0[j], qlane, 32);
            unsigned a1 = __shfl(d1[j], qlane, 32);
            pf.w[j] = hs ? a1 : a0;
            unsigned b0 = __shfl(d0[j], qlane + 16, 32);
            unsigned b1 = __shfl(d1[j], qlane + 16, 32);
            pf.w[4 + j] = hs ? b1 : b0;
        }

        Frag va0 = load_a(Vp + kj, TT);                    // d rows 0..15
        Frag va1 = load_a(Vp + (size_t)16 * TT + kj, TT);  // d rows 16..31
        olo = wmma16(va0, pf, olo);   // O^T[d][q] accumulation
        ohi = wmma16(va1, pf, ohi);
    }

    float inv = 1.f / lrun;
    int t = qt * 16 + qlane;
    union { unsigned w[4]; uint4 q; } plo, phi;
#pragma unroll
    for (int j = 0; j < 4; j++) {
        plo.w[j] = pack2(olo[2 * j] * inv, olo[2 * j + 1] * inv);
        phi.w[j] = pack2(ohi[2 * j] * inv, ohi[2 * j + 1] * inv);
    }
    _Float16* ob = O + ((size_t)(b * TT + t)) * HID + h * HD;
    *(uint4*)(ob + 8 * hs)      = plo.q;   // d = 8hs + 0..7
    *(uint4*)(ob + 16 + 8 * hs) = phi.q;   // d = 16 + 8hs + 0..7
}

// ---------------- Kernel 5: out-proj GEMM + bias + residual ----------------
__global__ void k_proj(const _Float16* __restrict__ wo,
                       const _Float16* __restrict__ O,
                       const float* __restrict__ bias,
                       const float* __restrict__ x,
                       float* __restrict__ y) {
    int task = blockIdx.x * 8 + (threadIdx.x >> 5);   // 16384 wave-tasks
    int tt = task & 255;
    int ot = (task >> 8) & 15;
    int b  = task >> 12;
    int obase = ot * 16, tbase = tt * 16;
    const _Float16* arow = wo + (size_t)obase * HID;
    const _Float16* brow = O + ((size_t)(b * TT + tbase)) * HID;
    v8f acc = {};
#pragma unroll
    for (int kc = 0; kc < HID; kc += 32) {
        Frag a  = load_a(arow + kc, HID);
        Frag bb = load_b(brow + kc, HID);
        acc = wmma16(a, bb, acc);
    }
    int lane = threadIdx.x & 31;
    int t  = tbase + (lane & 15);
    int o0 = obase + 8 * (lane >> 4);
#pragma unroll
    for (int r = 0; r < 8; r++) {
        int o = o0 + r;
        size_t idx = ((size_t)(b * C_IN + o)) * HW + t;
        y[idx] = acc[r] + bias[o] + x[idx];
    }
}

extern "C" void kernel_launch(void* const* d_in, const int* in_sizes, int n_in,
                              void* d_out, int out_size, void* d_ws, size_t ws_size,
                              hipStream_t stream) {
    (void)in_sizes; (void)n_in; (void)out_size; (void)ws_size;
    const float* x     = (const float*)d_in[0];
    const float* nw    = (const float*)d_in[1];
    const float* nb    = (const float*)d_in[2];
    const float* qkv_w = (const float*)d_in[3];
    const float* out_w = (const float*)d_in[4];
    const float* out_b = (const float*)d_in[5];
    float* y = (float*)d_out;

    char* ws = (char*)d_ws;
    size_t off = 0;
    _Float16* xnT = (_Float16*)(ws + off); off += (size_t)BATCH * HW * C_IN * 2;
    _Float16* wq  = (_Float16*)(ws + off); off += (size_t)384 * C_IN * 2;
    _Float16* wo  = (_Float16*)(ws + off); off += (size_t)C_IN * HID * 2;
    _Float16* Qb  = (_Float16*)(ws + off); off += (size_t)BATCH * NH * TT * HD * 2;
    _Float16* Kb  = (_Float16*)(ws + off); off += (size_t)BATCH * NH * TT * HD * 2;
    _Float16* Vt  = (_Float16*)(ws + off); off += (size_t)BATCH * NH * HD * TT * 2;
    _Float16* Ob  = (_Float16*)(ws + off); off += (size_t)BATCH * TT * HID * 2;

    k_groupnorm<<<32,   256, 0, stream>>>(x, nw, nb, xnT);
    k_cvt_w    <<<512,  256, 0, stream>>>(qkv_w, out_w, wq, wo);
    k_qkv      <<<3072, 256, 0, stream>>>(wq, xnT, Qb, Kb, Vt);
    k_attn     <<<512,  256, 0, stream>>>(Qb, Kb, Vt, Ob);
    k_proj     <<<2048, 256, 0, stream>>>(wo, Ob, out_b, x, y);
}
